// HGT_75617194213411
// MI455X (gfx1250) — matrix operand order repaired and checked by reading, not compiled, not run on hardware
//
#include <hip/hip_runtime.h>

typedef __attribute__((ext_vector_type(16))) __bf16 v16bf;
typedef __attribute__((ext_vector_type(8)))  __bf16 v8bf;
typedef __attribute__((ext_vector_type(16))) float  v16f;
typedef __attribute__((ext_vector_type(8)))  float  v8f;
typedef __attribute__((ext_vector_type(4)))  float  v4f;
typedef __attribute__((ext_vector_type(4)))  int    v4i;

static constexpr int NA   = 30000;
static constexpr int NP   = 60000;
static constexpr int NE   = 400000;
static constexpr int HIDC = 128;
static constexpr int NH   = 8;
static constexpr int OUTC = 16;

#ifndef __has_builtin
#define __has_builtin(x) 0
#endif
#if __has_builtin(__builtin_amdgcn_global_load_async_to_lds_b128) && \
    __has_builtin(__builtin_amdgcn_s_wait_asynccnt)
#define HAVE_ASYNC_LDS 1
#else
#define HAVE_ASYNC_LDS 0
#endif

// ---------------- float <-> ordered-uint encoding for atomic max ----------------
__device__ __forceinline__ unsigned fenc(float x) {
  unsigned u = __float_as_uint(x);
  return (u & 0x80000000u) ? ~u : (u | 0x80000000u);
}
__device__ __forceinline__ float fdec(unsigned u) {
  return __uint_as_float((u & 0x80000000u) ? (u & 0x7fffffffu) : ~u);
}

// ---------------- small prep kernels ----------------
// W [K][N] fp32 -> Wt [N][K] bf16
__global__ void transpose_to_bf16_kernel(const float* __restrict__ W,
                                         __bf16* __restrict__ Wt, int K, int N) {
  int i = blockIdx.x * blockDim.x + threadIdx.x;
  if (i >= K * N) return;
  int n = i % N, k = i / N;
  Wt[(size_t)n * K + k] = (__bf16)W[i];
}

// Wt[N=h*16+e][K=c] = sum_d W[c][h*16+d] * rel[h][d][e]   (fold per-head 16x16 into GEMM)
__global__ void fold_rel_kernel(const float* __restrict__ W, const float* __restrict__ rel,
                                __bf16* __restrict__ Wt) {
  int i = blockIdx.x * blockDim.x + threadIdx.x;
  if (i >= HIDC * HIDC) return;
  int he = i & (HIDC - 1), c = i >> 7;
  int h = he >> 4, e = he & 15;
  const float* wr = W + (size_t)c * HIDC + h * 16;
  const float* rr = rel + h * 256 + e;
  float acc = 0.f;
#pragma unroll
  for (int d = 0; d < 16; ++d) acc += wr[d] * rr[d * 16];
  Wt[(size_t)he * HIDC + c] = (__bf16)acc;
}

__global__ void fold_rel_bias_kernel(const float* __restrict__ b, const float* __restrict__ rel,
                                     float* __restrict__ bf) {
  int he = threadIdx.x;
  if (he >= HIDC) return;
  int h = he >> 4, e = he & 15;
  float acc = 0.f;
#pragma unroll
  for (int d = 0; d < 16; ++d) acc += b[h * 16 + d] * rel[h * 256 + d * 16 + e];
  bf[he] = acc;
}

__global__ void compute_beta_kernel(const float* sa, const float* sp, float* betas) {
  if (threadIdx.x == 0) {
    betas[0] = 1.f / (1.f + __expf(-sa[0]));
    betas[1] = 1.f / (1.f + __expf(-sp[0]));
  }
}

__global__ void fill_u32_kernel(unsigned* __restrict__ p, unsigned v, int n) {
  int i = blockIdx.x * blockDim.x + threadIdx.x;
  if (i < n) p[i] = v;
}

// ---------------- WMMA GEMM: C = epi(A @ Bt^T + bias) ----------------
// A fp32 [M][KK]; Bt bf16 [NN][KK] (pre-transposed weights, shared by all rows).
// B staged K-chunk-wise into LDS (gfx1250 async-copy path); each wave owns an
// RT x CT grid of 16x16 tiles so every B fragment feeds RT WMMAs and the whole
// fragment set stays register-resident (no per-WMMA dscnt stall).
// EPI: 0 = bias, 1 = bias+relu, 2 = beta*(acc+bias) + (1-beta)*resid
template <int NN, int CT, int EPI, int KK>
__global__ __launch_bounds__(256) void gemm_wmma_kernel(
    const float* __restrict__ A, const __bf16* __restrict__ Bt,
    const float* __restrict__ bias, const float* __restrict__ resid,
    const float* __restrict__ betas, int betaIdx,
    float* __restrict__ C, int M) {
  constexpr int CG  = NN / (16 * CT);   // waves spanning N
  constexpr int RG  = 8 / CG;           // row groups per block
  constexpr int RT  = 8 / RG;           // row tiles per wave
  constexpr int KCH = 128;              // K staged per round
  constexpr int SK  = KCH + 8;          // padded LDS row stride (16B skew)
  static_assert(CG * RG == 8 && RT * RG * 16 == 128, "bad blocking");
  static_assert(KK % KCH == 0, "K must be a multiple of 128");
  __shared__ __bf16 sB[NN * SK];

  const int lane = threadIdx.x & 31;
  const int wave = threadIdx.x >> 5;
  const int rg = wave / CG;
  const int cg = wave % CG;
  const int rowBase = blockIdx.x * 128 + rg * RT * 16;
  const bool active = rowBase < M;      // wave-uniform
  const int ml = lane & 15;
  const int kh = lane >> 4;             // which K-half this lane holds

  v8f acc[RT][CT];
#pragma unroll
  for (int i = 0; i < RT; ++i)
#pragma unroll
    for (int t = 0; t < CT; ++t)
#pragma unroll
      for (int r = 0; r < 8; ++r) acc[i][t][r] = 0.f;

  const float* arow[RT];
#pragma unroll
  for (int i = 0; i < RT; ++i) {
    int r = rowBase + i * 16 + ml;
    if (r >= M) r = M - 1;              // clamp: keeps loads in-bounds, values unused
    arow[i] = A + (size_t)r * KK;
  }

  for (int kc = 0; kc < KK; kc += KCH) {
    // ---- cooperative stage of B chunk [NN][KCH] into LDS ----
    constexpr int CPR = KCH / 8;        // 16B chunks per row
    for (int c = threadIdx.x; c < NN * CPR; c += 256) {
      int n = c / CPR, j = c % CPR;
      const __bf16* src = Bt + (size_t)n * KK + kc + j * 8;
      __bf16* dst = sB + n * SK + j * 8;
#if HAVE_ASYNC_LDS
      __builtin_amdgcn_global_load_async_to_lds_b128(
          (v4i*)src, (__attribute__((address_space(3))) v4i*)dst, 0, 0);
#else
      *(v8bf*)dst = *(const v8bf*)src;
#endif
    }
#if HAVE_ASYNC_LDS
    __builtin_amdgcn_s_wait_asynccnt(0);
#endif
    __syncthreads();

    if (active) {
#pragma unroll
      for (int kk = 0; kk < KCH; kk += 32) {
        const int k0 = kc + kk + kh * 8;
        v16bf a[RT];
#pragma unroll
        for (int i = 0; i < RT; ++i) {
          v8f g0 = *(const v8f*)(arow[i] + k0);        // K +0..7
          v8f g1 = *(const v8f*)(arow[i] + k0 + 16);   // K +16..23
          v16f full;
          full.lo = g0;
          full.hi = g1;
          a[i] = __builtin_convertvector(full, v16bf); // paired v_cvt_pk_bf16_f32
        }
        v16bf bf[CT];
#pragma unroll
        for (int t = 0; t < CT; ++t) {
          const __bf16* brow = sB + ((cg * CT + t) * 16 + ml) * SK + kk + kh * 8;
          bf[t].lo = *(const v8bf*)brow;
          bf[t].hi = *(const v8bf*)(brow + 16);
        }
#pragma unroll
        for (int i = 0; i < RT; ++i)
#pragma unroll
          for (int t = 0; t < CT; ++t)
            acc[i][t] = __builtin_amdgcn_wmma_f32_16x16x32_bf16(
                false, a[i], false, bf[t], (short)0, acc[i][t], false, false);
      }
    }
    __syncthreads();
  }

  if (!active) return;
  float beta = (EPI == 2) ? betas[betaIdx] : 0.f;
#pragma unroll
  for (int i = 0; i < RT; ++i) {
    if (rowBase + i * 16 >= M) continue;   // wave-uniform tail guard
#pragma unroll
    for (int t = 0; t < CT; ++t) {
      int n = (cg * CT + t) * 16 + ml;
      float bia = bias[n];
#pragma unroll
      for (int r = 0; r < 8; ++r) {
        size_t row = (size_t)rowBase + i * 16 + r + 8 * kh;
        float v = acc[i][t][r] + bia;
        if (EPI == 1) v = fmaxf(v, 0.f);
        if (EPI == 2) v = beta * v + (1.f - beta) * resid[row * (size_t)NN + n];
        C[row * (size_t)NN + n] = v;
      }
    }
  }
}

// ---------------- edge kernels ----------------
__global__ void edge_alpha_kernel(const float* __restrict__ q, const float* __restrict__ krel,
                                  const int* __restrict__ src, const int* __restrict__ dst,
                                  const float* __restrict__ pri, const float* __restrict__ vrel,
                                  float* __restrict__ alpha, unsigned* __restrict__ menc, int E) {
  int i = blockIdx.x * blockDim.x + threadIdx.x;
  if (i >= E * NH) return;
  int h = i & 7, e = i >> 3;
  int si = src[e], di = dst[e];
  const float* qp = q + (size_t)di * HIDC + h * 16;
  const float* kp = krel + (size_t)si * HIDC + h * 16;
  __builtin_prefetch(vrel + (size_t)si * HIDC + h * 16, 0, 1);  // warm L2 for the agg pass
  float acc = 0.f;
#pragma unroll
  for (int j = 0; j < 4; ++j) {
    v4f qv = *(const v4f*)(qp + 4 * j);
    v4f kv = *(const v4f*)(kp + 4 * j);
    acc += qv[0] * kv[0] + qv[1] * kv[1] + qv[2] * kv[2] + qv[3] * kv[3];
  }
  float a = acc * pri[h] * 0.25f;  // 1/sqrt(D=16)
  alpha[i] = a;
  atomicMax(menc + (size_t)di * NH + h, fenc(a));
}

__global__ void edge_agg_kernel(const float* __restrict__ alpha, const unsigned* __restrict__ menc,
                                const float* __restrict__ vrel, const int* __restrict__ src,
                                const int* __restrict__ dst, float* __restrict__ s,
                                float* __restrict__ agg, int E) {
  int i = blockIdx.x * blockDim.x + threadIdx.x;
  if (i >= E * NH) return;
  int h = i & 7, e = i >> 3;
  int si = src[e], di = dst[e];
  float mval = fdec(menc[(size_t)di * NH + h]);
  float w = __expf(alpha[i] - mval);
  atomicAdd(s + (size_t)di * NH + h, w);
  const float* vp = vrel + (size_t)si * HIDC + h * 16;
  float* ap = agg + (size_t)di * HIDC + h * 16;
#pragma unroll
  for (int j = 0; j < 4; ++j) {
    v4f vv = *(const v4f*)(vp + 4 * j);
    atomicAdd(ap + 4 * j + 0, w * vv[0]);
    atomicAdd(ap + 4 * j + 1, w * vv[1]);
    atomicAdd(ap + 4 * j + 2, w * vv[2]);
    atomicAdd(ap + 4 * j + 3, w * vv[3]);
  }
}

__global__ void norm_gelu_kernel(float* __restrict__ agg, const float* __restrict__ s, int n) {
  int i = blockIdx.x * blockDim.x + threadIdx.x;
  if (i >= n * HIDC) return;
  int c = i & (HIDC - 1);
  int node = i >> 7;
  int h = c >> 4;
  float denom = fmaxf(s[(size_t)node * NH + h], 1e-16f);
  float x = agg[i] / denom;
  float t = tanhf(0.7978845608028654f * (x + 0.044715f * x * x * x));
  agg[i] = 0.5f * x * (1.f + t);
}

#define GEMM(NNv, CTv, EPIv, KKv, A_, B_, bias_, resid_, betas_, bidx_, C_, M_)            \
  gemm_wmma_kernel<NNv, CTv, EPIv, KKv><<<dim3(((M_) + 127) / 128), dim3(256), 0, stream>>>( \
      A_, B_, bias_, resid_, betas_, bidx_, C_, M_)

extern "C" void kernel_launch(void* const* d_in, const int* in_sizes, int n_in,
                              void* d_out, int out_size, void* d_ws, size_t ws_size,
                              hipStream_t stream) {
  (void)in_sizes; (void)n_in; (void)out_size; (void)ws_size;
  // ---- inputs (jax sorted-dict pytree flatten order) ----
  const float* x_author = (const float*)d_in[0];
  const float* x_paper  = (const float*)d_in[1];
  const float* inW_a = (const float*)d_in[2];
  const float* inb_a = (const float*)d_in[3];
  const float* inW_p = (const float*)d_in[4];
  const float* inb_p = (const float*)d_in[5];
  // per-layer 24-entry sorted block starting at 6 + 24*l:
  // 0 a_W_author 1 a_W_paper 2 a_b_author 3 a_b_paper 4 arel_wb 5 arel_writes
  // 6 k_W_author 7 k_W_paper 8 k_b_author 9 k_b_paper 10 mrel_wb 11 mrel_writes
  // 12 pri_wb 13 pri_writes 14 q_W_author 15 q_W_paper 16 q_b_author 17 q_b_paper
  // 18 skip_author 19 skip_paper 20 v_W_author 21 v_W_paper 22 v_b_author 23 v_b_paper
  const float* outW = (const float*)d_in[54];
  const float* outb = (const float*)d_in[55];
  const int* wsrc = (const int*)d_in[56];
  const int* wdst = (const int*)d_in[57];
  const int* bsrc = (const int*)d_in[58];
  const int* bdst = (const int*)d_in[59];

  // ---- workspace layout (fp32 region then bf16 weight region) ----
  float* base = (float*)d_ws;
  size_t off = 0;
  auto alloc = [&](size_t n) { float* p = base + off; off += n; return p; };
  float* xa[2]  = {alloc((size_t)NA * HIDC), alloc((size_t)NA * HIDC)};
  float* xp[2]  = {alloc((size_t)NP * HIDC), alloc((size_t)NP * HIDC)};
  float* qbuf   = alloc((size_t)NP * HIDC);
  float* krel   = alloc((size_t)NP * HIDC);
  float* vrel   = alloc((size_t)NP * HIDC);
  float* agg    = alloc((size_t)NP * HIDC);
  float* alpha  = alloc((size_t)NE * NH);
  float* sbuf   = alloc((size_t)NP * NH);
  unsigned* menc = (unsigned*)alloc((size_t)NP * NH);
  float* bk     = alloc(HIDC);
  float* bv     = alloc(HIDC);
  float* betas  = alloc(4);
  __bf16* bw = (__bf16*)(base + off);
  size_t boff = 0;
  auto balloc = [&](size_t n) { __bf16* p = bw + boff; boff += n; return p; };
  __bf16* WtInA = balloc(128 * 128);
  __bf16* WtInP = balloc(256 * 128);
  __bf16* WtOut = balloc(128 * 16);
  __bf16* WtQ   = balloc(128 * 128);
  __bf16* WtK   = balloc(128 * 128);
  __bf16* WtV   = balloc(128 * 128);
  __bf16* WtA   = balloc(128 * 128);

  const unsigned NEG_INF_ENC = 0x007FFFFFu;  // fenc(-inf)
  dim3 blk(256);
  auto grid1 = [](long long n) { return dim3((unsigned)((n + 255) / 256)); };

  // ---- input projections (relu epilogue) ----
  transpose_to_bf16_kernel<<<grid1(128 * 128), blk, 0, stream>>>(inW_a, WtInA, 128, 128);
  transpose_to_bf16_kernel<<<grid1(256 * 128), blk, 0, stream>>>(inW_p, WtInP, 256, 128);
  transpose_to_bf16_kernel<<<grid1(128 * 16), blk, 0, stream>>>(outW, WtOut, 128, 16);
  GEMM(128, 4, 1, 128, x_author, WtInA, inb_a, nullptr, nullptr, 0, xa[0], NA);
  GEMM(128, 4, 1, 256, x_paper,  WtInP, inb_p, nullptr, nullptr, 0, xp[0], NP);

  int cur = 0;
  for (int l = 0; l < 2; ++l) {
    const int b0 = 6 + 24 * l;
    auto P = [&](int k) { return (const float*)d_in[b0 + k]; };
    int nxt = 1 - cur;
    compute_beta_kernel<<<1, 32, 0, stream>>>(P(18), P(19), betas);

    // ===== relation "writes": author -> paper =====
    transpose_to_bf16_kernel<<<grid1(128 * 128), blk, 0, stream>>>(P(15), WtQ, 128, 128); // q_W_paper
    fold_rel_kernel<<<grid1(128 * 128), blk, 0, stream>>>(P(6), P(5), WtK);   // k_W_author x arel_writes
    fold_rel_bias_kernel<<<1, 128, 0, stream>>>(P(8), P(5), bk);
    fold_rel_kernel<<<grid1(128 * 128), blk, 0, stream>>>(P(20), P(11), WtV); // v_W_author x mrel_writes
    fold_rel_bias_kernel<<<1, 128, 0, stream>>>(P(22), P(11), bv);
    GEMM(128, 4, 0, 128, xp[cur], WtQ, P(17), nullptr, nullptr, 0, qbuf, NP);
    GEMM(128, 4, 0, 128, xa[cur], WtK, bk, nullptr, nullptr, 0, krel, NA);
    GEMM(128, 4, 0, 128, xa[cur], WtV, bv, nullptr, nullptr, 0, vrel, NA);
    fill_u32_kernel<<<grid1((long long)NP * NH), blk, 0, stream>>>(menc, NEG_INF_ENC, NP * NH);
    (void)hipMemsetAsync(sbuf, 0, (size_t)NP * NH * 4, stream);
    (void)hipMemsetAsync(agg, 0, (size_t)NP * HIDC * 4, stream);
    edge_alpha_kernel<<<grid1((long long)NE * NH), blk, 0, stream>>>(qbuf, krel, wsrc, wdst, P(13), vrel, alpha, menc, NE);
    edge_agg_kernel<<<grid1((long long)NE * NH), blk, 0, stream>>>(alpha, menc, vrel, wsrc, wdst, sbuf, agg, NE);
    norm_gelu_kernel<<<grid1((long long)NP * HIDC), blk, 0, stream>>>(agg, sbuf, NP);
    transpose_to_bf16_kernel<<<grid1(128 * 128), blk, 0, stream>>>(P(1), WtA, 128, 128);   // a_W_paper
    GEMM(128, 4, 2, 128, agg, WtA, P(3), xp[cur], betas, 1, xp[nxt], NP);

    // ===== relation "wb": paper -> author =====
    transpose_to_bf16_kernel<<<grid1(128 * 128), blk, 0, stream>>>(P(14), WtQ, 128, 128); // q_W_author
    fold_rel_kernel<<<grid1(128 * 128), blk, 0, stream>>>(P(7), P(4), WtK);   // k_W_paper x arel_wb
    fold_rel_bias_kernel<<<1, 128, 0, stream>>>(P(9), P(4), bk);
    fold_rel_kernel<<<grid1(128 * 128), blk, 0, stream>>>(P(21), P(10), WtV); // v_W_paper x mrel_wb
    fold_rel_bias_kernel<<<1, 128, 0, stream>>>(P(23), P(10), bv);
    GEMM(128, 4, 0, 128, xa[cur], WtQ, P(16), nullptr, nullptr, 0, qbuf, NA);
    GEMM(128, 4, 0, 128, xp[cur], WtK, bk, nullptr, nullptr, 0, krel, NP);
    GEMM(128, 4, 0, 128, xp[cur], WtV, bv, nullptr, nullptr, 0, vrel, NP);
    fill_u32_kernel<<<grid1((long long)NA * NH), blk, 0, stream>>>(menc, NEG_INF_ENC, NA * NH);
    (void)hipMemsetAsync(sbuf, 0, (size_t)NA * NH * 4, stream);
    (void)hipMemsetAsync(agg, 0, (size_t)NA * HIDC * 4, stream);
    edge_alpha_kernel<<<grid1((long long)NE * NH), blk, 0, stream>>>(qbuf, krel, bsrc, bdst, P(12), vrel, alpha, menc, NE);
    edge_agg_kernel<<<grid1((long long)NE * NH), blk, 0, stream>>>(alpha, menc, vrel, bsrc, bdst, sbuf, agg, NE);
    norm_gelu_kernel<<<grid1((long long)NA * HIDC), blk, 0, stream>>>(agg, sbuf, NA);
    transpose_to_bf16_kernel<<<grid1(128 * 128), blk, 0, stream>>>(P(0), WtA, 128, 128);   // a_W_author
    GEMM(128, 4, 2, 128, agg, WtA, P(2), xa[cur], betas, 0, xa[nxt], NA);

    cur = nxt;
  }

  // ---- classifier ----
  GEMM(16, 1, 0, 128, xa[cur], WtOut, outb, nullptr, nullptr, 0, (float*)d_out, NA);
}
